// DualPathGeomAttention_67697274520077
// MI455X (gfx1250) — matrix operand order repaired and checked by reading, not compiled, not run on hardware
//
#include <hip/hip_runtime.h>
#include <math.h>

typedef __attribute__((ext_vector_type(16))) __bf16 v16bf;
typedef __attribute__((ext_vector_type(8)))  float  v8f;

#define B_ 2
#define L_ 2048
#define S_ 2048
#define H_ 8
#define E_ 64
#define D_ 64
#define EPS_ 1e-8f
#define SCALE_ 0.125f          // 1/sqrt(64)
#define SCHUNK 32
#define ROWS_PER_WAVE 16
#define WAVES 8
#define BLOCK_ROWS (ROWS_PER_WAVE * WAVES)   // 128

// row reductions: a matrix row lives in one 16-lane half (ISA C/D layout),
// xor masks 1..8 stay inside the half
__device__ __forceinline__ float redmax16(float v) {
  v = fmaxf(v, __shfl_xor(v, 1, 32));
  v = fmaxf(v, __shfl_xor(v, 2, 32));
  v = fmaxf(v, __shfl_xor(v, 4, 32));
  v = fmaxf(v, __shfl_xor(v, 8, 32));
  return v;
}
__device__ __forceinline__ float redsum16(float v) {
  v += __shfl_xor(v, 1, 32);
  v += __shfl_xor(v, 2, 32);
  v += __shfl_xor(v, 4, 32);
  v += __shfl_xor(v, 8, 32);
  return v;
}

__global__ void init_ws_kernel(float* ws) {
  ws[0] = 0.f;
  ws[1] = 0.f;
}

__global__ void finalize_reg_kernel(const float* ws, float* reg_out) {
  const float inv_cnt = 1.0f / (float)((size_t)B_ * H_ * L_ * S_);
  reg_out[0] = 0.5f * (ws[0] * inv_cnt + ws[1] * inv_cnt);
}

__global__ __launch_bounds__(256) void geom_attn_main(
    const float* __restrict__ Q, const float* __restrict__ K,
    const float* __restrict__ Vd, const float* __restrict__ Vw,
    const float* __restrict__ fg, float* __restrict__ out,
    float* __restrict__ regsum)
{
  __shared__ __bf16 KL[32][64];          // K chunk, row-major (e contiguous)
  __shared__ __bf16 VdT[64][32];         // V chunks transposed (s contiguous)
  __shared__ __bf16 VwT[64][32];
  __shared__ float  kn2s[32];
  __shared__ float  qn2s[WAVES][16];
  __shared__ __bf16 PLd[WAVES][16][32];  // per-wave prob tiles (dot)
  __shared__ __bf16 PLw[WAVES][16][32];  // per-wave prob tiles (wedge)
  __shared__ float  s_regd, s_regw;

  const int tid  = threadIdx.x;
  const int lane = tid & 31;
  const int wv   = tid >> 5;
  const int h    = lane >> 4;   // lane half
  const int n    = lane & 15;   // col-in-tile (B/C/D) and row (A)

  const int lb = blockIdx.x & 15;        // L/128 = 16 row blocks
  const int bh = blockIdx.x >> 4;        // B*H = 16
  const int b  = bh >> 3;
  const int hh = bh & 7;
  const int l0 = lb * BLOCK_ROWS + wv * ROWS_PER_WAVE;

  // ---- Q A-fragments (16x32 bf16 each, contiguous k = 16h + s mapping) ----
  v16bf aQ0, aQ1;
  {
    const float* qr = Q + (((size_t)b * L_ + (l0 + n)) * H_ + hh) * E_;
#pragma unroll
    for (int i = 0; i < 8; ++i) {
      const int kb = h * 16 + i * 2;
      aQ0[2*i]   = (__bf16)qr[kb];
      aQ0[2*i+1] = (__bf16)qr[kb + 1];
      aQ1[2*i]   = (__bf16)qr[32 + kb];
      aQ1[2*i+1] = (__bf16)qr[32 + kb + 1];
    }
  }
  if (lane < 16) {   // |q|^2 per row of this wave's tile
    const float* qr = Q + (((size_t)b * L_ + (l0 + lane)) * H_ + hh) * E_;
    float acc = 0.f;
#pragma unroll
    for (int e = 0; e < E_; ++e) acc += qr[e] * qr[e];
    qn2s[wv][lane] = acc;
  }
  if (tid == 0) { s_regd = 0.f; s_regw = 0.f; }
  __syncthreads();

  float qn2v[8];
#pragma unroll
  for (int r = 0; r < 8; ++r) qn2v[r] = qn2s[wv][r + 8 * h];

  // ---- running state: 4 N-tiles x (rows r+8h) per lane, two paths ----
  // lD/lW are PER-LANE PARTIAL row sums (this lane's 2 columns per chunk);
  // the rescale factor f is lane-uniform within a half, so the linear
  // recurrence commutes with the final cross-lane reduction in the epilogue.
  v8f accD[4] = {};
  v8f accW[4] = {};
  float mD[8], lD[8], mW[8], lW[8];
#pragma unroll
  for (int r = 0; r < 8; ++r) { mD[r] = -3.0e38f; lD[r] = 0.f; mW[r] = -3.0e38f; lW[r] = 0.f; }

  float regd = 0.f, regw = 0.f;

  for (int s0 = 0; s0 < S_; s0 += SCHUNK) {
    __syncthreads();                       // previous chunk fully consumed
    // ---- cooperative stage: f32 -> bf16 into LDS ----
    for (int idx = tid; idx < SCHUNK * 64; idx += 256) {
      const int s = idx >> 6, e = idx & 63;
      const size_t rowb = ((size_t)b * S_ + s0 + s) * H_ + hh;
      KL[s][e]  = (__bf16)K [rowb * E_ + e];
      VdT[e][s] = (__bf16)Vd[rowb * D_ + e];
      VwT[e][s] = (__bf16)Vw[rowb * D_ + e];
    }
    if (tid < SCHUNK) {                    // |k|^2 per chunk column
      const float* kr = K + (((size_t)b * S_ + s0 + tid) * H_ + hh) * E_;
      float acc = 0.f;
#pragma unroll
      for (int e = 0; e < E_; ++e) acc += kr[e] * kr[e];
      kn2s[tid] = acc;
      if (s0 + SCHUNK < S_)                // global_prefetch_b8 of next K chunk
        __builtin_prefetch(K + (((size_t)b * S_ + s0 + SCHUNK + tid) * H_ + hh) * E_, 0, 1);
    }
    __syncthreads();

    // ---- scores: two 16x16 tiles via 4 WMMAs ----
    v8f d0 = {}, d1 = {};
    {
      v16bf bk;
#pragma unroll
      for (int i = 0; i < 8; ++i) { const int k = h*16 + 2*i;
        bk[2*i] = KL[n][k]; bk[2*i+1] = KL[n][k+1]; }
      d0 = __builtin_amdgcn_wmma_f32_16x16x32_bf16(false, aQ0, false, bk, (short)0, d0, false, false);
#pragma unroll
      for (int i = 0; i < 8; ++i) { const int k = h*16 + 2*i;
        bk[2*i] = KL[n][32+k]; bk[2*i+1] = KL[n][32+k+1]; }
      d0 = __builtin_amdgcn_wmma_f32_16x16x32_bf16(false, aQ1, false, bk, (short)0, d0, false, false);
#pragma unroll
      for (int i = 0; i < 8; ++i) { const int k = h*16 + 2*i;
        bk[2*i] = KL[16+n][k]; bk[2*i+1] = KL[16+n][k+1]; }
      d1 = __builtin_amdgcn_wmma_f32_16x16x32_bf16(false, aQ0, false, bk, (short)0, d1, false, false);
#pragma unroll
      for (int i = 0; i < 8; ++i) { const int k = h*16 + 2*i;
        bk[2*i] = KL[16+n][32+k]; bk[2*i+1] = KL[16+n][32+k+1]; }
      d1 = __builtin_amdgcn_wmma_f32_16x16x32_bf16(false, aQ1, false, bk, (short)0, d1, false, false);
    }

    // ---- dual online softmax (dot + wedge) ----
    const float k20 = kn2s[n], k21 = kn2s[16 + n];
    float facD[8], facW[8];
#pragma unroll
    for (int r = 0; r < 8; ++r) {
      const float dr0 = d0[r], dr1 = d1[r];
      const float sd0 = dr0 * SCALE_, sd1 = dr1 * SCALE_;
      const float w0 = sqrtf(fmaxf(qn2v[r] * k20 - dr0 * dr0, 0.f) + EPS_) * SCALE_;
      const float w1 = sqrtf(fmaxf(qn2v[r] * k21 - dr1 * dr1, 0.f) + EPS_) * SCALE_;
      regd += fabsf(sd0) + fabsf(sd1);
      regw += w0 + w1;                     // wedge scores are non-negative

      float cm = redmax16(fmaxf(sd0, sd1));
      float mn = fmaxf(mD[r], cm);
      float f  = __expf(mD[r] - mn);
      float p0 = __expf(sd0 - mn), p1 = __expf(sd1 - mn);
      lD[r] = lD[r] * f + (p0 + p1);       // per-lane partial row sum
      mD[r] = mn; facD[r] = f;
      PLd[wv][r + 8*h][n]      = (__bf16)p0;
      PLd[wv][r + 8*h][16 + n] = (__bf16)p1;

      cm = redmax16(fmaxf(w0, w1));
      mn = fmaxf(mW[r], cm);
      f  = __expf(mW[r] - mn);
      p0 = __expf(w0 - mn); p1 = __expf(w1 - mn);
      lW[r] = lW[r] * f + (p0 + p1);       // per-lane partial row sum
      mW[r] = mn; facW[r] = f;
      PLw[wv][r + 8*h][n]      = (__bf16)p0;
      PLw[wv][r + 8*h][16 + n] = (__bf16)p1;
    }
#pragma unroll
    for (int nt = 0; nt < 4; ++nt)
#pragma unroll
      for (int r = 0; r < 8; ++r) { accD[nt][r] *= facD[r]; accW[nt][r] *= facW[r]; }

    asm volatile("s_wait_dscnt 0" ::: "memory");   // same-wave LDS RAW on P tiles

    // ---- P A-fragments from per-wave LDS tiles ----
    v16bf aPd, aPw;
#pragma unroll
    for (int i = 0; i < 8; ++i) {
      const int kb = h * 16 + i * 2;
      aPd[2*i]   = PLd[wv][n][kb];
      aPd[2*i+1] = PLd[wv][n][kb + 1];
      aPw[2*i]   = PLw[wv][n][kb];
      aPw[2*i+1] = PLw[wv][n][kb + 1];
    }
    // ---- P @ V: 4 N-tiles x 2 paths = 8 WMMAs ----
#pragma unroll
    for (int nt = 0; nt < 4; ++nt) {
      v16bf bv;
#pragma unroll
      for (int i = 0; i < 8; ++i) { const int k = h*16 + 2*i;
        bv[2*i] = VdT[nt*16 + n][k]; bv[2*i+1] = VdT[nt*16 + n][k+1]; }
      accD[nt] = __builtin_amdgcn_wmma_f32_16x16x32_bf16(false, aPd, false, bv, (short)0, accD[nt], false, false);
#pragma unroll
      for (int i = 0; i < 8; ++i) { const int k = h*16 + 2*i;
        bv[2*i] = VwT[nt*16 + n][k]; bv[2*i+1] = VwT[nt*16 + n][k+1]; }
      accW[nt] = __builtin_amdgcn_wmma_f32_16x16x32_bf16(false, aPw, false, bv, (short)0, accW[nt], false, false);
    }
  }

  // ---- epilogue: reduce partial row sums once, normalize, fuse, store ----
  float lDt[8], lWt[8];
#pragma unroll
  for (int r = 0; r < 8; ++r) {
    lDt[r] = redsum16(lD[r]);
    lWt[r] = redsum16(lW[r]);
  }
  const float g = 1.f / (1.f + __expf(-fg[0]));
#pragma unroll
  for (int nt = 0; nt < 4; ++nt) {
#pragma unroll
    for (int r = 0; r < 8; ++r) {
      const int row = l0 + r + 8 * h;
      const int d   = nt * 16 + n;
      const float od = accD[nt][r] / lDt[r];
      const float ow = accW[nt][r] / lWt[r];
      out[(((size_t)b * L_ + row) * H_ + hh) * D_ + d] = (1.f - g) * od + g * ow;
    }
  }

  atomicAdd(&s_regd, regd);
  atomicAdd(&s_regw, regw);
  __syncthreads();
  if (tid == 0) {
    atomicAdd(&regsum[0], s_regd);
    atomicAdd(&regsum[1], s_regw);
  }
}

extern "C" void kernel_launch(void* const* d_in, const int* in_sizes, int n_in,
                              void* d_out, int out_size, void* d_ws, size_t ws_size,
                              hipStream_t stream) {
  const float* Q  = (const float*)d_in[0];
  const float* K  = (const float*)d_in[1];
  const float* Vd = (const float*)d_in[2];
  const float* Vw = (const float*)d_in[3];
  const float* fg = (const float*)d_in[4];
  float* out = (float*)d_out;
  float* ws  = (float*)d_ws;

  init_ws_kernel<<<1, 1, 0, stream>>>(ws);
  geom_attn_main<<<dim3(B_ * H_ * (L_ / BLOCK_ROWS)), dim3(256), 0, stream>>>(
      Q, K, Vd, Vw, fg, out, ws);
  finalize_reg_kernel<<<1, 1, 0, stream>>>(ws, out + (size_t)B_ * L_ * H_ * D_);
}
